// FMoELinearProj_86002425135819
// MI455X (gfx1250) — matrix operand order repaired and checked by reading, not compiled, not run on hardware
//
#include <hip/hip_runtime.h>

// MoE grouped GEMM + low-rank projection, fused for MI455X (gfx1250).
//   res[n, j] = sum_i inp[n,i] * WfT[e(n)][j][i] + bproj[e(n)][j]
// where WfT[e][j][i] = sum_d W[e][d][i] * Cmat[d][j]  (Cmat[d][k*64+s] = c_pinv[k][d][s])
// and   bproj[e][j]  = sum_d b[e][d] * Cmat[d][j].
// All matrix math via v_wmma_f32_16x16x32_bf16 (bf16 in, f32 accumulate).

typedef __attribute__((ext_vector_type(16))) __bf16   v16bf;
typedef __attribute__((ext_vector_type(2)))  __bf16   v2bf;
typedef __attribute__((ext_vector_type(8)))  float    v8f;
typedef __attribute__((ext_vector_type(4)))  float    f32x4;
typedef __attribute__((ext_vector_type(4)))  unsigned u32x4;

union BF16Frag {
  v16bf    v;
  unsigned u[8];
  u32x4    q[2];
};

// Pack two f32 -> dword of two bf16. Lowered by the backend to a single
// native v_cvt_pk_bf16_f32 on gfx1250 (verified in round-3 disasm).
__device__ __forceinline__ unsigned pk(float lo, float hi) {
  v2bf t;
  t.x = (__bf16)lo;
  t.y = (__bf16)hi;
  return __builtin_bit_cast(unsigned, t);
}

// ---------------------------------------------------------------------------
// Kernel 1: bproj[e][j] = sum_d bias[e][d] * c_pinv[j/64][d][j%64]   (tiny)
// ---------------------------------------------------------------------------
__global__ __launch_bounds__(256) void bias_proj_kernel(
    const float* __restrict__ bias, const float* __restrict__ c_pinv,
    float* __restrict__ bproj) {
  int gid = blockIdx.x * 256 + threadIdx.x;   // 16 * 1024
  int e = gid >> 10, j = gid & 1023;
  int ke = j >> 6, s = j & 63;
  const float* b = bias + e * 1024;
  const float* c = c_pinv + (size_t)ke * 65536 + s;
  float acc = 0.f;
  for (int d = 0; d < 1024; ++d) acc = fmaf(b[d], c[(size_t)d * 64], acc);
  bproj[gid] = acc;
}

// ---------------------------------------------------------------------------
// Kernel 2: WfT[e][j][i] (bf16) = sum_d W[e][d][i] * Cmat[d][j]
// GEMM with M=i, N=j, K=d. Both operands contracted over their slow (row)
// dim, so stage 32x32 f32 tiles in LDS (b128 in / b128 out, 36-float row
// pitch keeps quads 16B-aligned and columns conflict-light) and read
// transposed per the WMMA A/B lane layouts. Block = 4 waves, 32x32 tile.
// ---------------------------------------------------------------------------
__global__ __launch_bounds__(128) void fuse_weights_wmma(
    const float* __restrict__ weight, const float* __restrict__ c_pinv,
    unsigned short* __restrict__ wfT) {
  __shared__ float Ws[32][36];   // Ws[d][i] tile of W[e]
  __shared__ float Cs[32][36];   // Cs[d][j] tile of Cmat

  const int it = blockIdx.x;     // i tile (32 wide)
  const int jt = blockIdx.y;     // j tile (32 wide)
  const int e  = blockIdx.z;

  const int lane = threadIdx.x;
  const int wave = threadIdx.y;             // 0..3
  const int wm = wave >> 1, wn = wave & 1;  // 2x2 wave grid of 16x16 subtiles
  const int ln = lane & 15, half = lane >> 4;
  const int ml = wm * 16 + ln;              // local i (A row)
  const int nl = wn * 16 + ln;              // local j (B col)
  const int tid = wave * 32 + lane;         // 0..127

  const int ke = jt >> 1;                   // (jt*32)/64 : constant per block
  const int s0 = (jt & 1) * 32;
  const float* wBase = weight + (size_t)e * 1024 * 1024 + it * 32;
  const float* cBase = c_pinv + (size_t)ke * 65536 + s0;

  // Staging assignment: 256 quads (32 rows x 8 quads) over 128 threads.
  const int q0r = tid >> 3;                 // rows 0..15   (pass 0)
  const int q0c = (tid & 7) * 4;
  v8f acc = {};

  for (int d0 = 0; d0 < 1024; d0 += 32) {
    __syncthreads();
#pragma unroll
    for (int n = 0; n < 2; ++n) {           // uniform trip count, b128 moves
      int r = q0r + n * 16;
      *(f32x4*)&Ws[r][q0c] = *(const f32x4*)&wBase[(size_t)(d0 + r) * 1024 + q0c];
      *(f32x4*)&Cs[r][q0c] = *(const f32x4*)&cBase[(size_t)(d0 + r) * 64 + q0c];
    }
    __syncthreads();

    BF16Frag a, b;
#pragma unroll
    for (int v = 0; v < 4; ++v) {
      int kk = half * 8 + 2 * v;            // A layout: two 8-runs per lane
      a.u[v]     = pk(Ws[kk][ml],      Ws[kk + 1][ml]);
      a.u[v + 4] = pk(Ws[kk + 16][ml], Ws[kk + 17][ml]);
      int kb = half * 8 + 2 * v;            // B layout: two 8-runs per lane
      b.u[v]     = pk(Cs[kb][nl],      Cs[kb + 1][nl]);
      b.u[v + 4] = pk(Cs[kb + 16][nl], Cs[kb + 17][nl]);
    }
    acc = __builtin_amdgcn_wmma_f32_16x16x32_bf16(
        false, a.v, false, b.v, (short)0, acc, false, false);
  }

  // D(m=i, n=j); store transposed -> WfT[j][i]: per-lane 8 consecutive i
  // values = one 16-byte bf16 store.
  const int j  = jt * 32 + wn * 16 + ln;
  const int ib = it * 32 + wm * 16 + 8 * half;
  u32x4 o;
  o.x = pk(acc[0], acc[1]); o.y = pk(acc[2], acc[3]);
  o.z = pk(acc[4], acc[5]); o.w = pk(acc[6], acc[7]);
  *(u32x4*)(wfT + ((size_t)(e * 1024 + j)) * 1024 + ib) = o;
}

// ---------------------------------------------------------------------------
// Kernel 3: main grouped GEMM.  out[row][j] = bproj[e][j] + inp[row] . WfT[e][j]
// Block = 8 waves (32x8), block tile 128 rows x 64 cols; wave tile 16x64
// (one A frag feeds 4 WMMAs). A loaded straight from global f32 (contiguous
// 32B runs per lane), B loaded as pre-packed bf16 b128s. No LDS in hot loop.
// ---------------------------------------------------------------------------
__global__ __launch_bounds__(256) void moe_proj_wmma(
    const float* __restrict__ inp, const int* __restrict__ cnt,
    const unsigned short* __restrict__ wfT, const float* __restrict__ bproj,
    float* __restrict__ out) {
  const int jt = blockIdx.x;     // 0..15 : 64-col output tile
  const int tt = blockIdx.y;     // token tile within expert
  const int e  = blockIdx.z;     // expert

  int off = 0;
#pragma unroll
  for (int q = 0; q < 16; ++q) off += (q < e) ? cnt[q] : 0;
  const int count = cnt[e];
  if (tt * 128 >= count) return;            // uniform early exit
  const int row0   = off + tt * 128;
  const int rowEnd = off + count;

  const int lane = threadIdx.x;
  const int wave = threadIdx.y;             // 0..7 -> 16 rows each
  const int ln = lane & 15, half = lane >> 4;
  const int rowBase = row0 + wave * 16;

  v8f acc[4];
#pragma unroll
  for (int f = 0; f < 4; ++f) {
    float bv = bproj[e * 1024 + jt * 64 + f * 16 + ln];
#pragma unroll
    for (int g = 0; g < 8; ++g) acc[f][g] = bv;
  }

  int rowA = rowBase + ln;                  // A-fragment row for this lane
  if (rowA >= rowEnd) rowA = rowEnd - 1;    // clamp (masked at store)
  const float* aRow = inp + (size_t)rowA * 1024;
  const unsigned short* bBase =
      wfT + ((size_t)(e * 1024 + jt * 64 + ln)) * 1024 + half * 16;

#pragma unroll 2
  for (int k0 = 0; k0 < 1024; k0 += 32) {
    BF16Frag a;
    const f32x4* p0 = (const f32x4*)(aRow + k0 + half * 16);
    f32x4 r0 = p0[0], r1 = p0[1];
    const f32x4* p1 = (const f32x4*)(aRow + k0 + 8 + half * 16);
    f32x4 r2 = p1[0], r3 = p1[1];
    a.u[0] = pk(r0.x, r0.y); a.u[1] = pk(r0.z, r0.w);
    a.u[2] = pk(r1.x, r1.y); a.u[3] = pk(r1.z, r1.w);
    a.u[4] = pk(r2.x, r2.y); a.u[5] = pk(r2.z, r2.w);
    a.u[6] = pk(r3.x, r3.y); a.u[7] = pk(r3.z, r3.w);
    __builtin_prefetch((const void*)(aRow + k0 + 64), 0, 1);  // global_prefetch_b8
#pragma unroll
    for (int f = 0; f < 4; ++f) {
      BF16Frag b;
      const u32x4* pb = (const u32x4*)(bBase + (size_t)f * 16 * 1024 + k0);
      b.q[0] = pb[0]; b.q[1] = pb[1];
      acc[f] = __builtin_amdgcn_wmma_f32_16x16x32_bf16(
          false, a.v, false, b.v, (short)0, acc[f], false, false);
    }
  }

  // D layout: lane -> n = ln, vgpr g -> m = g + 8*half.
  float* oBase = out + jt * 64 + ln;
#pragma unroll
  for (int g = 0; g < 8; ++g) {
    int row = rowBase + g + 8 * half;
    if (row < rowEnd) {
      float* o = oBase + (size_t)row * 1024;
      o[0]  = acc[0][g];
      o[16] = acc[1][g];
      o[32] = acc[2][g];
      o[48] = acc[3][g];
    }
  }
}

// ---------------------------------------------------------------------------
extern "C" void kernel_launch(void* const* d_in, const int* in_sizes, int n_in,
                              void* d_out, int out_size, void* d_ws, size_t ws_size,
                              hipStream_t stream) {
  (void)in_sizes; (void)n_in; (void)out_size; (void)ws_size;
  const float* inp    = (const float*)d_in[0];
  const int*   cnt    = (const int*)d_in[1];
  const float* weight = (const float*)d_in[2];
  const float* bias   = (const float*)d_in[3];
  const float* c_pinv = (const float*)d_in[4];
  float* out = (float*)d_out;

  // Workspace: [0, 64KB) bproj f32 ; [64KB, 64KB+32MB) WfT bf16
  float*          bproj = (float*)d_ws;
  unsigned short* wfT   = (unsigned short*)((char*)d_ws + (64 << 10));

  bias_proj_kernel<<<64, 256, 0, stream>>>(bias, c_pinv, bproj);
  fuse_weights_wmma<<<dim3(32, 32, 16), dim3(32, 4), 0, stream>>>(weight, c_pinv, wfT);
  moe_proj_wmma<<<dim3(16, 512, 16), dim3(32, 8), 0, stream>>>(inp, cnt, wfT, bproj, out);
}
